// TwoStageAttentionLayer_custom_9895604650601
// MI455X (gfx1250) — matrix operand order repaired and checked by reading, not compiled, not run on hardware
//
#include <hip/hip_runtime.h>
#include <math.h>

// ---------------------------------------------------------------------------
// TwoStageAttentionLayer for MI455X (gfx1250, wave32, WMMA).
// All large matmuls run on v_wmma_f32_16x16x32_f16 (f16 in, f32 accumulate).
// Weights are pre-converted once per call to transposed f16 (Wt16[N][K]) so
// GEMM B-staging is a pure 16B f16 copy; A tiles are staged with b128 loads +
// packed f32->f16 conversion. GEMM main loop is double-buffered in LDS.
// Stage-1 attention is fused (QK^T -> dist-weighted softmax -> PV) per (b,h);
// the reference's *global* softmax max cancels in the normalization, so the
// per-row max is mathematically identical.
// ---------------------------------------------------------------------------

typedef __attribute__((ext_vector_type(16))) _Float16 v16h;
typedef __attribute__((ext_vector_type(8)))  _Float16 h8;
typedef __attribute__((ext_vector_type(4)))  _Float16 h4;
typedef __attribute__((ext_vector_type(8)))  float    v8f;
typedef __attribute__((ext_vector_type(4)))  float    f4v;

#define WMMA16(a, b, c) \
  __builtin_amdgcn_wmma_f32_16x16x32_f16(false, (a), false, (b), (short)0, (c), false, false)

__device__ __forceinline__ v16h pack16(h8 lo, h8 hi) {
  return __builtin_shufflevector(lo, hi, 0,1,2,3,4,5,6,7,8,9,10,11,12,13,14,15);
}

// ---------------------------------------------------------------------------
// Weight pre-pass: O[n*K + k] = (f16) W[k*N + n], 32x32 LDS-tiled transpose.
// grid = (N/32, K/32), block = 256.
// ---------------------------------------------------------------------------
__global__ __launch_bounds__(256)
void k_cvt_w(const float* __restrict__ W, _Float16* __restrict__ O, int K, int N)
{
  __shared__ _Float16 t[32][33];
  const int kb = blockIdx.y * 32, nb = blockIdx.x * 32;
  const int tid = threadIdx.x;
  #pragma unroll
  for (int i = 0; i < 4; ++i) {
    int e = i * 256 + tid;            // 0..1023
    int r = e >> 5, c = e & 31;       // r: k-offset, c: n-offset
    t[c][r] = (_Float16)W[(size_t)(kb + r) * N + nb + c];
  }
  __syncthreads();
  #pragma unroll
  for (int i = 0; i < 4; ++i) {
    int e = i * 256 + tid;
    int r = e >> 5, c = e & 31;       // r: n-offset, c: k-offset
    O[(size_t)(nb + r) * K + kb + c] = t[r][c];
  }
}

// ---------------------------------------------------------------------------
// GEMM: C[M,N] = act(A[M,K] @ W[K,N] + bias[N]), W given as Wt16[N][K] f16.
// M%128==0, N%128==0, K%32==0. ACT: 0 none, 1 exact GELU, 2 (tanh+1)/2.
// 256 threads = 8 waves; block tile 128x128; wave tile 32x64; K step 32;
// double-buffered LDS with register prefetch of the next tile.
// ---------------------------------------------------------------------------
template <int ACT>
__global__ __launch_bounds__(256)
void k_gemm(const float* __restrict__ A, const _Float16* __restrict__ Wt16,
            const float* __restrict__ bias, float* __restrict__ C,
            int M, int N, int K)
{
  __shared__ _Float16 Ah[2][128][40];   // [buf][m][k], pitch 40 halfs (80B)
  __shared__ _Float16 Wt[2][128][40];   // [buf][n][k]

  const int tid  = threadIdx.x;
  const int lane = tid & 31, wv = tid >> 5;
  const int hi   = lane >> 4, ln = lane & 15;
  const int m0   = blockIdx.y * 128, n0 = blockIdx.x * 128;
  const int wm   = wv & 3, wn = wv >> 2;   // 4 waves along M, 2 along N
  const int S    = K >> 5;

  v8f acc[2][4];
  #pragma unroll
  for (int a = 0; a < 2; ++a)
    #pragma unroll
    for (int b = 0; b < 4; ++b)
      acc[a][b] = (v8f){0.f,0.f,0.f,0.f,0.f,0.f,0.f,0.f};

  f4v areg[4];
  h8  wreg[2];

  auto loadG = [&](int kt) {
    #pragma unroll
    for (int i = 0; i < 4; ++i) {
      int f = i * 256 + tid;            // 0..1023 float4 chunks of A tile
      int r = f >> 3, c4 = f & 7;
      areg[i] = *(const f4v*)&A[(size_t)(m0 + r) * K + kt + c4 * 4];
    }
    #pragma unroll
    for (int i = 0; i < 2; ++i) {
      int e = i * 256 + tid;            // 0..511 h8 chunks of W tile
      int r = e >> 2, c8 = e & 3;
      wreg[i] = *(const h8*)&Wt16[(size_t)(n0 + r) * K + kt + c8 * 8];
    }
  };
  auto storeL = [&](int buf) {
    #pragma unroll
    for (int i = 0; i < 4; ++i) {
      int f = i * 256 + tid;
      int r = f >> 3, c4 = f & 7;
      *(h4*)&Ah[buf][r][c4 * 4] = __builtin_convertvector(areg[i], h4);
    }
    #pragma unroll
    for (int i = 0; i < 2; ++i) {
      int e = i * 256 + tid;
      int r = e >> 2, c8 = e & 3;
      *(h8*)&Wt[buf][r][c8 * 8] = wreg[i];
    }
  };

  loadG(0);
  storeL(0);
  __syncthreads();

  for (int s = 0; s < S; ++s) {
    if (s + 1 < S) loadG((s + 1) * 32);   // global prefetch overlaps WMMAs
    const int cur = s & 1;
    #pragma unroll
    for (int mt = 0; mt < 2; ++mt) {
      const _Float16* ap = &Ah[cur][wm * 32 + mt * 16 + ln][hi * 8];
      v16h a = pack16(*(const h8*)ap, *(const h8*)(ap + 16));
      #pragma unroll
      for (int nt = 0; nt < 4; ++nt) {
        const _Float16* bp = &Wt[cur][wn * 64 + nt * 16 + ln][hi * 8];
        v16h b = pack16(*(const h8*)bp, *(const h8*)(bp + 16));
        acc[mt][nt] = WMMA16(a, b, acc[mt][nt]);
      }
    }
    if (s + 1 < S) {
      storeL(1 - cur);
      __syncthreads();
    }
  }

  #pragma unroll
  for (int mt = 0; mt < 2; ++mt) {
    #pragma unroll
    for (int nt = 0; nt < 4; ++nt) {
      int col = n0 + wn * 64 + nt * 16 + ln;
      float bb = bias ? bias[col] : 0.f;
      #pragma unroll
      for (int r = 0; r < 8; ++r) {
        int row = m0 + wm * 32 + mt * 16 + r + 8 * hi;
        float v = acc[mt][nt][r] + bb;
        if (ACT == 1)      v = 0.5f * v * (1.0f + erff(v * 0.7071067811865476f));
        else if (ACT == 2) v = 0.5f * (tanhf(v) + 1.0f);
        C[(size_t)row * N + col] = v;
      }
    }
  }
}

// ---------------------------------------------------------------------------
// Fused stage-1 attention. One block (256 thr = 8 waves) per (b, h),
// grid = 2048. Q/K/V are [32768,256] (row b*128+l, col h*32+e).
// D (distance weights, already (tanh+1)/2) is [32768,1024] (col h*128+s).
// ---------------------------------------------------------------------------
__global__ __launch_bounds__(256)
void k_attn1(const float* __restrict__ Q, const float* __restrict__ Kb,
             const float* __restrict__ Vb, const float* __restrict__ D,
             float* __restrict__ O)
{
  __shared__ _Float16 Qh[128][40];    // [l][e]
  __shared__ _Float16 Kh[128][40];    // [s][e]  (== B^T layout for QK^T)
  __shared__ _Float16 Vt[32][136];    // [e][s]  (== B^T layout for P.V)
  __shared__ _Float16 P [128][136];   // normalized probs, f16

  const int tid  = threadIdx.x;
  const int lane = tid & 31, wv = tid >> 5;
  const int hi   = lane >> 4, ln = lane & 15;
  const int b    = blockIdx.x >> 3, h = blockIdx.x & 7;
  const size_t base = ((size_t)b * 128) * 256 + (size_t)h * 32;

  #pragma unroll
  for (int i = 0; i < 4; ++i) {
    int f = i * 256 + tid;            // 0..1023 float4 chunks
    int r = f >> 3, c4 = f & 7;
    size_t g = base + (size_t)r * 256 + c4 * 4;
    f4v q = *(const f4v*)&Q[g];
    f4v k = *(const f4v*)&Kb[g];
    f4v v = *(const f4v*)&Vb[g];
    *(h4*)&Qh[r][c4 * 4] = __builtin_convertvector(q, h4);
    *(h4*)&Kh[r][c4 * 4] = __builtin_convertvector(k, h4);
    h4 hv = __builtin_convertvector(v, h4);
    #pragma unroll
    for (int j = 0; j < 4; ++j) Vt[c4 * 4 + j][r] = hv[j];
  }
  __syncthreads();

  // ---- S = Q K^T : wave wv owns rows 16*wv..16*wv+15, all 128 cols ----
  v8f acc[8];
  #pragma unroll
  for (int j = 0; j < 8; ++j) acc[j] = (v8f){0.f,0.f,0.f,0.f,0.f,0.f,0.f,0.f};

  {
    const _Float16* ap = &Qh[wv * 16 + ln][hi * 8];
    v16h a = pack16(*(const h8*)ap, *(const h8*)(ap + 16));
    #pragma unroll
    for (int j = 0; j < 8; ++j) {
      const _Float16* bp = &Kh[j * 16 + ln][hi * 8];
      v16h bf = pack16(*(const h8*)bp, *(const h8*)(bp + 16));
      acc[j] = WMMA16(a, bf, acc[j]);
    }
  }

  // ---- distance-weighted softmax, per row, registers + shuffles ----
  const float scale = 0.17677669529663687f;  // 1/sqrt(32)
  #pragma unroll
  for (int r = 0; r < 8; ++r) {
    const int grow = wv * 16 + r + 8 * hi;
    float m = -3.0e38f;
    #pragma unroll
    for (int j = 0; j < 8; ++j) m = fmaxf(m, acc[j][r]);
    #pragma unroll
    for (int msk = 1; msk <= 8; msk <<= 1) m = fmaxf(m, __shfl_xor(m, msk, 32));
    float w[8];
    float sum = 0.f;
    #pragma unroll
    for (int j = 0; j < 8; ++j) {
      int col = j * 16 + ln;
      float d = D[((size_t)b * 128 + grow) * 1024 + (size_t)h * 128 + col];
      float e = __expf((acc[j][r] - m) * scale) * d;
      w[j] = e;
      sum += e;
    }
    #pragma unroll
    for (int msk = 1; msk <= 8; msk <<= 1) sum += __shfl_xor(sum, msk, 32);
    float inv = 1.0f / sum;
    #pragma unroll
    for (int j = 0; j < 8; ++j)
      P[grow][j * 16 + ln] = (_Float16)(w[j] * inv);
  }
  __syncthreads();

  // ---- O = P V ----
  v8f o[2];
  o[0] = (v8f){0.f,0.f,0.f,0.f,0.f,0.f,0.f,0.f};
  o[1] = (v8f){0.f,0.f,0.f,0.f,0.f,0.f,0.f,0.f};
  #pragma unroll
  for (int kk = 0; kk < 4; ++kk) {
    const _Float16* ap = &P[wv * 16 + ln][kk * 32 + hi * 8];
    v16h a = pack16(*(const h8*)ap, *(const h8*)(ap + 16));
    #pragma unroll
    for (int nt = 0; nt < 2; ++nt) {
      const _Float16* bp = &Vt[nt * 16 + ln][kk * 32 + hi * 8];
      v16h bf = pack16(*(const h8*)bp, *(const h8*)(bp + 16));
      o[nt] = WMMA16(a, bf, o[nt]);
    }
  }
  #pragma unroll
  for (int nt = 0; nt < 2; ++nt)
    #pragma unroll
    for (int r = 0; r < 8; ++r)
      O[base + (size_t)(wv * 16 + r + 8 * hi) * 256 + nt * 16 + ln] = o[nt][r];
}

// ---------------------------------------------------------------------------
// Residual + LayerNorm over 256-wide rows. One wave per row, 8 rows / block.
// ---------------------------------------------------------------------------
__global__ __launch_bounds__(256)
void k_ln(const float* __restrict__ X, const float* __restrict__ R,
          const float* __restrict__ g, const float* __restrict__ be,
          float* __restrict__ Y)
{
  const int lane = threadIdx.x & 31, wv = threadIdx.x >> 5;
  const size_t row = (size_t)blockIdx.x * 8 + wv;
  const float* xr = X + row * 256;
  const float* rr = R + row * 256;
  f4v v[2];
  float s = 0.f;
  #pragma unroll
  for (int i = 0; i < 2; ++i) {
    int c = lane * 4 + i * 128;
    v[i] = *(const f4v*)&xr[c] + *(const f4v*)&rr[c];
    s += v[i][0] + v[i][1] + v[i][2] + v[i][3];
  }
  #pragma unroll
  for (int m = 1; m < 32; m <<= 1) s += __shfl_xor(s, m, 32);
  const float mean = s * (1.0f / 256.0f);
  float q = 0.f;
  #pragma unroll
  for (int i = 0; i < 2; ++i)
    #pragma unroll
    for (int j = 0; j < 4; ++j) { float d = v[i][j] - mean; q += d * d; }
  #pragma unroll
  for (int m = 1; m < 32; m <<= 1) q += __shfl_xor(q, m, 32);
  const float rstd = rsqrtf(q * (1.0f / 256.0f) + 1e-5f);
  #pragma unroll
  for (int i = 0; i < 2; ++i) {
    int c = lane * 4 + i * 128;
    f4v gg = *(const f4v*)&g[c];
    f4v bb = *(const f4v*)&be[c];
    f4v o;
    #pragma unroll
    for (int j = 0; j < 4; ++j) o[j] = (v[i][j] - mean) * rstd * gg[j] + bb[j];
    *(f4v*)&Y[row * 256 + c] = o;
  }
}

// ---------------------------------------------------------------------------
// Row permutes (float4 copies, 64 float4 per 256-float row).
// ---------------------------------------------------------------------------
__global__ __launch_bounds__(256)
void k_perm_send(const float4* __restrict__ in, float4* __restrict__ out)
{
  int t = blockIdx.x * 256 + threadIdx.x;
  int q = t & 63, orow = t >> 6;
  int d = orow & 31, bl = orow >> 5;
  int l = bl & 127, b = bl >> 7;
  int irow = (b * 32 + d) * 128 + l;
  out[(size_t)orow * 64 + q] = in[(size_t)irow * 64 + q];
}

__global__ __launch_bounds__(256)
void k_perm_out(const float4* __restrict__ in, float4* __restrict__ out)
{
  int t = blockIdx.x * 256 + threadIdx.x;
  int q = t & 63, orow = t >> 6;
  int l = orow & 127, bd = orow >> 7;
  int d = bd & 31, b = bd >> 5;
  int irow = (b * 128 + l) * 32 + d;
  out[(size_t)orow * 64 + q] = in[(size_t)irow * 64 + q];
}

// ---------------------------------------------------------------------------
// Stage-2 "send": 10 router queries attend over D=32 keys. One wave/(bl,h).
// ---------------------------------------------------------------------------
__global__ __launch_bounds__(32)
void k_attn_send(const float* __restrict__ Qr, const float* __restrict__ Kb,
                 const float* __restrict__ Vb, float* __restrict__ O)
{
  __shared__ float Ks[32][33], Vs[32][33];
  const int lane = threadIdx.x;
  const int bl = blockIdx.x >> 3, h = blockIdx.x & 7;
  const int l = bl & 127;
  for (int j = 0; j < 32; ++j) {
    Ks[j][lane] = Kb[((size_t)bl * 32 + j) * 256 + h * 32 + lane];
    Vs[j][lane] = Vb[((size_t)bl * 32 + j) * 256 + h * 32 + lane];
  }
  __syncthreads();
  const float scale = 0.17677669529663687f;
  for (int i = 0; i < 10; ++i) {
    const float* q = Qr + ((size_t)l * 10 + i) * 256 + h * 32;
    float s = 0.f;
    #pragma unroll
    for (int e = 0; e < 32; ++e) s += q[e] * Ks[lane][e];
    s *= scale;
    float m = s;
    #pragma unroll
    for (int msk = 1; msk < 32; msk <<= 1) m = fmaxf(m, __shfl_xor(m, msk, 32));
    float w = __expf(s - m);
    float sum = w;
    #pragma unroll
    for (int msk = 1; msk < 32; msk <<= 1) sum += __shfl_xor(sum, msk, 32);
    float a = w / sum;
    float accv = 0.f;
    #pragma unroll
    for (int j = 0; j < 32; ++j) accv += __shfl(a, j, 32) * Vs[j][lane];
    O[((size_t)bl * 10 + i) * 256 + h * 32 + lane] = accv;
  }
}

// ---------------------------------------------------------------------------
// Stage-2 "recv": 32 queries attend over 10 keys. One wave/(bl,h).
// ---------------------------------------------------------------------------
__global__ __launch_bounds__(32)
void k_attn_recv(const float* __restrict__ Qb, const float* __restrict__ Kb,
                 const float* __restrict__ Vb, float* __restrict__ O)
{
  __shared__ float Ks[10][33], Vs[10][33];
  const int lane = threadIdx.x;
  const int bl = blockIdx.x >> 3, h = blockIdx.x & 7;
  for (int idx = lane; idx < 320; idx += 32) {
    int j = idx >> 5, e = idx & 31;
    Ks[j][e] = Kb[((size_t)bl * 10 + j) * 256 + h * 32 + e];
    Vs[j][e] = Vb[((size_t)bl * 10 + j) * 256 + h * 32 + e];
  }
  __syncthreads();
  const float scale = 0.17677669529663687f;
  float q[32];
  #pragma unroll
  for (int e = 0; e < 32; ++e)
    q[e] = Qb[((size_t)bl * 32 + lane) * 256 + h * 32 + e];
  float sc[10];
  float m = -3.0e38f;
  #pragma unroll
  for (int j = 0; j < 10; ++j) {
    float s = 0.f;
    #pragma unroll
    for (int e = 0; e < 32; ++e) s += q[e] * Ks[j][e];
    sc[j] = s * scale;
    m = fmaxf(m, sc[j]);
  }
  float sum = 0.f;
  #pragma unroll
  for (int j = 0; j < 10; ++j) { sc[j] = __expf(sc[j] - m); sum += sc[j]; }
  const float inv = 1.0f / sum;
  #pragma unroll
  for (int e = 0; e < 32; ++e) {
    float accv = 0.f;
    #pragma unroll
    for (int j = 0; j < 10; ++j) accv += sc[j] * Vs[j][e];
    O[((size_t)bl * 32 + lane) * 256 + h * 32 + e] = accv * inv;
  }
}

// ---------------------------------------------------------------------------
// Host orchestration. Input order = setup_inputs() dict insertion order.
// ---------------------------------------------------------------------------
extern "C" void kernel_launch(void* const* d_in, const int* in_sizes, int n_in,
                              void* d_out, int out_size, void* d_ws, size_t ws_size,
                              hipStream_t stream)
{
  (void)in_sizes; (void)n_in; (void)out_size; (void)ws_size;

  const float* x      = (const float*)d_in[0];
  const float* tWq    = (const float*)d_in[1];
  const float* tbq    = (const float*)d_in[2];
  const float* tWk    = (const float*)d_in[3];
  const float* tbk    = (const float*)d_in[4];
  const float* tWv    = (const float*)d_in[5];
  const float* tbv    = (const float*)d_in[6];
  const float* tWo    = (const float*)d_in[7];
  const float* tbo    = (const float*)d_in[8];
  const float* tWqd   = (const float*)d_in[9];
  const float* tbqd   = (const float*)d_in[10];
  const float* sWq    = (const float*)d_in[11];
  const float* sbq    = (const float*)d_in[12];
  const float* sWk    = (const float*)d_in[13];
  const float* sbk    = (const float*)d_in[14];
  const float* sWv    = (const float*)d_in[15];
  const float* sbv    = (const float*)d_in[16];
  const float* sWo    = (const float*)d_in[17];
  const float* sbo    = (const float*)d_in[18];
  const float* rWq    = (const float*)d_in[19];
  const float* rbq    = (const float*)d_in[20];
  const float* rWk    = (const float*)d_in[21];
  const float* rbk    = (const float*)d_in[22];
  const float* rWv    = (const float*)d_in[23];
  const float* rbv    = (const float*)d_in[24];
  const float* rWo    = (const float*)d_in[25];
  const float* rbo    = (const float*)d_in[26];
  const float* router = (const float*)d_in[27];
  const float* ln1g   = (const float*)d_in[28];
  const float* ln1b   = (const float*)d_in[29];
  const float* ln2g   = (const float*)d_in[30];
  const float* ln2b   = (const float*)d_in[31];
  const float* ln3g   = (const float*)d_in[32];
  const float* ln3b   = (const float*)d_in[33];
  const float* ln4g   = (const float*)d_in[34];
  const float* ln4b   = (const float*)d_in[35];
  const float* m1W1   = (const float*)d_in[36];
  const float* m1b1   = (const float*)d_in[37];
  const float* m1W2   = (const float*)d_in[38];
  const float* m1b2   = (const float*)d_in[39];
  const float* m2W1   = (const float*)d_in[40];
  const float* m2b1   = (const float*)d_in[41];
  const float* m2W2   = (const float*)d_in[42];
  const float* m2b2   = (const float*)d_in[43];

  float* ws = (float*)d_ws;
  const size_t U    = 32768ull * 256;    // 8.39M floats
  const size_t BIGW = 32768ull * 1024;   // 33.55M floats
  float* BIG = ws;                        // dist / mlp1 hidden / mlp2 hidden
  float* b0  = ws + BIGW;
  float* b1  = b0 + U;
  float* b2  = b1 + U;
  float* b3  = b2 + U;
  float* b4  = b3 + U;
  float* b5  = b4 + U;
  float* sm0 = b5 + U;                    // routerQ  [1280,256]
  float* sm1 = sm0 + 1280ull * 256;       // send out [10240,256]
  float* sm2 = sm1 + 10240ull * 256;      // dim_buffer
  float* sm3 = sm2 + 10240ull * 256;      // recvK
  float* sm4 = sm3 + 10240ull * 256;      // recvV

  // f16 transposed weight arena (4 MB)
  _Float16* w16  = (_Float16*)(sm4 + 10240ull * 256);
  _Float16* wtq  = w16;              _Float16* wtk  = wtq  + 65536;
  _Float16* wtv  = wtk  + 65536;     _Float16* wto  = wtv  + 65536;
  _Float16* wtqd = wto  + 65536;     // 256x1024
  _Float16* wsq  = wtqd + 262144;    _Float16* wsk  = wsq  + 65536;
  _Float16* wsv  = wsk  + 65536;     _Float16* wso  = wsv  + 65536;
  _Float16* wrq  = wso  + 65536;     _Float16* wrk  = wrq  + 65536;
  _Float16* wrv  = wrk  + 65536;     _Float16* wro  = wrv  + 65536;
  _Float16* wm1a = wro  + 65536;     // 256x1024
  _Float16* wm1b = wm1a + 262144;    // 1024x256
  _Float16* wm2a = wm1b + 262144;
  _Float16* wm2b = wm2a + 262144;

  const dim3 blk(256);
  auto CVT = [&](const float* W, _Float16* O, int K, int N) {
    k_cvt_w<<<dim3(N / 32, K / 32), blk, 0, stream>>>(W, O, K, N);
  };
  auto GEMM = [&](int act, const float* A, const _Float16* W, const float* bi,
                  float* C, int M, int N, int K) {
    dim3 g(N / 128, M / 128);
    if (act == 0)      k_gemm<0><<<g, blk, 0, stream>>>(A, W, bi, C, M, N, K);
    else if (act == 1) k_gemm<1><<<g, blk, 0, stream>>>(A, W, bi, C, M, N, K);
    else               k_gemm<2><<<g, blk, 0, stream>>>(A, W, bi, C, M, N, K);
  };

  // ---------------- weight conversion pre-pass ----------------------------
  CVT(tWq,  wtq,  256, 256);  CVT(tWk,  wtk,  256, 256);
  CVT(tWv,  wtv,  256, 256);  CVT(tWo,  wto,  256, 256);
  CVT(tWqd, wtqd, 256, 1024);
  CVT(sWq,  wsq,  256, 256);  CVT(sWk,  wsk,  256, 256);
  CVT(sWv,  wsv,  256, 256);  CVT(sWo,  wso,  256, 256);
  CVT(rWq,  wrq,  256, 256);  CVT(rWk,  wrk,  256, 256);
  CVT(rWv,  wrv,  256, 256);  CVT(rWo,  wro,  256, 256);
  CVT(m1W1, wm1a, 256, 1024); CVT(m1W2, wm1b, 1024, 256);
  CVT(m2W1, wm2a, 256, 1024); CVT(m2W2, wm2b, 1024, 256);

  // ---------------- Stage 1: time attention (custom softmax) --------------
  GEMM(0, x, wtq,  tbq,  b0,  32768, 256,  256);   // Q
  GEMM(0, x, wtk,  tbk,  b1,  32768, 256,  256);   // K
  GEMM(0, x, wtv,  tbv,  b2,  32768, 256,  256);   // V
  GEMM(2, x, wtqd, tbqd, BIG, 32768, 1024, 256);   // dist = (tanh(.)+1)/2
  k_attn1<<<2048, 256, 0, stream>>>(b0, b1, b2, BIG, b3);
  GEMM(0, b3, wto, tbo, b4, 32768, 256, 256);      // time_enc
  k_ln<<<4096, 256, 0, stream>>>(x, b4, ln1g, ln1b, b5);   // dim_in
  GEMM(1, b5,  wm1a, m1b1, BIG, 32768, 1024, 256); // GELU(mlp1)
  GEMM(0, BIG, wm1b, m1b2, b4,  32768, 256, 1024);
  k_ln<<<4096, 256, 0, stream>>>(b5, b4, ln2g, ln2b, b0);  // dim_in (post-mlp)

  // ---------------- Stage 2: router dimension attention -------------------
  k_perm_send<<<8192, 256, 0, stream>>>((const float4*)b0, (float4*)b1); // dim_send
  GEMM(0, router, wsq, sbq, sm0, 1280,  256, 256); // router Q (shared over b)
  GEMM(0, b1,     wsk, sbk, b2,  32768, 256, 256); // send K
  GEMM(0, b1,     wsv, sbv, b3,  32768, 256, 256); // send V
  k_attn_send<<<8192, 32, 0, stream>>>(sm0, b2, b3, sm1);
  GEMM(0, sm1, wso, sbo, sm2, 10240, 256, 256);    // dim_buffer
  GEMM(0, b1,  wrq, rbq, b4,  32768, 256, 256);    // recv Q
  GEMM(0, sm2, wrk, rbk, sm3, 10240, 256, 256);    // recv K
  GEMM(0, sm2, wrv, rbv, sm4, 10240, 256, 256);    // recv V
  k_attn_recv<<<8192, 32, 0, stream>>>(b4, sm3, sm4, b0);
  GEMM(0, b0, wro, rbo, b4, 32768, 256, 256);      // dim_receive
  k_ln<<<4096, 256, 0, stream>>>(b1, b4, ln3g, ln3b, b0);  // dim_enc
  GEMM(1, b0,  wm2a, m2b1, BIG, 32768, 1024, 256); // GELU(mlp2)
  GEMM(0, BIG, wm2b, m2b2, b4,  32768, 256, 1024);
  k_ln<<<4096, 256, 0, stream>>>(b0, b4, ln4g, ln4b, b2);  // final dim_enc

  // ---------------- Output permute [B,L,D,dm] -> [B,D,L,dm] ----------------
  k_perm_out<<<8192, 256, 0, stream>>>((const float4*)b2, (float4*)d_out);
}
